// DecoderLayer_78735340471041
// MI455X (gfx1250) — compile-verified, matrix operand order checked by reading, not run on hardware
//
#include <hip/hip_runtime.h>
#include <hip/hip_bf16.h>

// ---------------- problem constants ----------------
constexpr int Bb   = 8;
constexpr int Tt   = 512;
constexpr int Ss   = 512;
constexpr int Dd   = 1024;
constexpr int Hh   = 16;
constexpr int DH   = 64;      // Dd / Hh
constexpr int Ee   = 8;
constexpr int Ff   = 4096;
constexpr int CAPc = 640;
constexpr int NTOK = Bb * Tt; // 4096

// ---------------- types ----------------
typedef __attribute__((ext_vector_type(16))) __bf16         v16bf;
typedef __attribute__((ext_vector_type(8)))  float          v8f;
typedef __attribute__((ext_vector_type(8)))  unsigned short ushort8;

union BF16x16 { v16bf v; unsigned short s[16]; };

__device__ __forceinline__ unsigned short f2bf(float f) {
  unsigned int u = __float_as_uint(f);
  unsigned int r = u + 0x7FFFu + ((u >> 16) & 1u);   // round-to-nearest-even
  return (unsigned short)(r >> 16);
}

__device__ __forceinline__ void ldfrag(BF16x16& f, const unsigned short* p) {
  *(ushort8*)&f.s[0] = *(const ushort8*)p;
  *(ushort8*)&f.s[8] = *(const ushort8*)(p + 16);
}

__device__ __forceinline__ float rowmax16(float v) {
#pragma unroll
  for (int m = 1; m < 16; m <<= 1) v = fmaxf(v, __shfl_xor(v, m, 32));
  return v;
}
__device__ __forceinline__ float rowsum16(float v) {
#pragma unroll
  for (int m = 1; m < 16; m <<= 1) v += __shfl_xor(v, m, 32);
  return v;
}

// ---------------- TDM (guarded; 5-arg on clang-22 / 6-arg on clang>=23) ----------------
#if __has_builtin(__builtin_amdgcn_tensor_load_to_lds) && __has_builtin(__builtin_amdgcn_s_wait_tensorcnt)
#define HAVE_TDM 1
#else
#define HAVE_TDM 0
#endif

#if HAVE_TDM
typedef __attribute__((ext_vector_type(4))) unsigned int u32x4;
typedef __attribute__((ext_vector_type(8))) int          i32x8;
typedef __attribute__((ext_vector_type(4))) int          i32x4;

// 2D bf16 tile load: tile_k (contig elems) x tile_rows, row stride = stride_elems.
__device__ __forceinline__ void tdm_load_2d(unsigned lds_off, const unsigned short* gp,
                                            unsigned tile_k, unsigned tile_rows,
                                            unsigned tensor_k, unsigned tensor_rows,
                                            unsigned long long stride_elems) {
  unsigned long long ga = (unsigned long long)gp;
  u32x4 g0 = { 1u,                                        // count=1, user-mode D#
               lds_off,                                   // lds_addr
               (unsigned)(ga & 0xFFFFFFFFu),              // global_addr[31:0]
               (unsigned)(((ga >> 32) & 0x01FFFFFFu) | 0x80000000u) }; // addr[56:32] | type=2
  i32x8 g1 = { (int)0x00010000u,                          // data_size=1 (2 bytes)
               (int)((tensor_k & 0xFFFFu) << 16),         // tensor_dim0[15:0] @bit48
               (int)((tensor_k >> 16) | ((tensor_rows & 0xFFFFu) << 16)),
               (int)((tensor_rows >> 16) | (tile_k << 16)),  // tile_dim0 @bit112
               (int)tile_rows,                            // tile_dim1 (tile_dim2=0 -> 2D)
               (int)(unsigned)(stride_elems & 0xFFFFFFFFull),
               (int)(unsigned)((stride_elems >> 32) & 0xFFFFull),
               0 };
  i32x4 z4 = { 0, 0, 0, 0 };
#if __clang_major__ >= 23
  i32x8 z8 = { 0, 0, 0, 0, 0, 0, 0, 0 };
  __builtin_amdgcn_tensor_load_to_lds(g0, g1, z4, z4, z8, 0);
#else
  __builtin_amdgcn_tensor_load_to_lds(g0, g1, z4, z4, 0);
#endif
}
#endif

// ---------------- weight prep ----------------
__global__ void dl_conv_bf16(const float* __restrict__ a, unsigned short* __restrict__ o, long n) {
  long i = (long)blockIdx.x * blockDim.x + threadIdx.x;
  if (i < n) o[i] = f2bf(a[i]);
}

// a: [E][R][C] f32  ->  o: [E][C][R] bf16   (R,C multiples of 32)
__global__ void dl_transconv(const float* __restrict__ a, unsigned short* __restrict__ o,
                             int R, int C) {
  __shared__ float t[32][33];
  long eoff = (long)blockIdx.z * R * C;
  int r0 = blockIdx.y * 32, c0 = blockIdx.x * 32;
  int tx = threadIdx.x & 31, ty = threadIdx.x >> 5;   // 32 x 8
#pragma unroll
  for (int i = 0; i < 32; i += 8)
    t[ty + i][tx] = a[eoff + (long)(r0 + ty + i) * C + (c0 + tx)];
  __syncthreads();
#pragma unroll
  for (int i = 0; i < 32; i += 8)
    o[eoff + (long)(c0 + ty + i) * R + (r0 + tx)] = f2bf(t[tx][ty + i]);
}

// ---------------- layernorm (writes f32 and bf16 copies) ----------------
__global__ void dl_ln(const float* __restrict__ x, const float* __restrict__ w,
                      const float* __restrict__ b, float* __restrict__ y,
                      unsigned short* __restrict__ ybf) {
  int row = blockIdx.x, tid = threadIdx.x;
  const float* xr = x + (long)row * Dd;
  float s = 0.f, s2 = 0.f;
  for (int d = tid; d < Dd; d += blockDim.x) { float v = xr[d]; s += v; s2 += v * v; }
#pragma unroll
  for (int m = 1; m < 32; m <<= 1) { s += __shfl_xor(s, m, 32); s2 += __shfl_xor(s2, m, 32); }
  __shared__ float sh[8], sh2[8];
  int wid = tid >> 5, ln = tid & 31;
  if (ln == 0) { sh[wid] = s; sh2[wid] = s2; }
  __syncthreads();
  if (tid == 0) {
    float a = 0.f, a2 = 0.f;
    for (int i = 0; i < (int)(blockDim.x >> 5); ++i) { a += sh[i]; a2 += sh2[i]; }
    sh[0] = a; sh2[0] = a2;
  }
  __syncthreads();
  float mean = sh[0] / (float)Dd;
  float var  = sh2[0] / (float)Dd - mean * mean;
  float inv  = rsqrtf(var + 1e-5f);
  for (int d = tid; d < Dd; d += blockDim.x) {
    float v = (xr[d] - mean) * inv * w[d] + b[d];
    y[(long)row * Dd + d] = v;
    ybf[(long)row * Dd + d] = f2bf(v);
  }
}

// ---------------- C-tile epilogue ----------------
__device__ __forceinline__ void store_tile(v8f acc, int row0, int col, long ldc,
                                           const float* bias, const float* Res,
                                           float* Cf, unsigned short* Cbf, int relu) {
  float bv = bias ? bias[col] : 0.f;
#pragma unroll
  for (int r = 0; r < 8; ++r) {
    float v = acc[r] + bv;
    if (relu) v = fmaxf(v, 0.f);
    long idx = (long)(row0 + r) * ldc + col;
    if (Res) v += Res[idx];
    if (Cf)  Cf[idx] = v;
    else     Cbf[idx] = f2bf(v);
  }
}

// ---------------- register-blocked WMMA GEMM (32x32 per wave) ----------------
// C[M,N] = act(A[M,K] @ Bt[N,K]^T + bias) (+residual). A bf16, Bt bf16 (row n contiguous in K).
__global__ void dl_gemm_wmma(const unsigned short* __restrict__ Ab, long lda,
                             const unsigned short* __restrict__ Bt,
                             const float* __restrict__ bias,
                             float* __restrict__ Cf, unsigned short* __restrict__ Cbf,
                             long ldc, const float* __restrict__ Res,
                             int M, int N, int K, int relu) {
  int wid  = (int)(((long)blockIdx.x * blockDim.x + threadIdx.x) >> 5);
  int lane = threadIdx.x & 31;
  int ntn  = N >> 5;
  int tiles = (M >> 5) * ntn;
  if (wid >= tiles) return;                       // wave-uniform
  int tm = wid / ntn, tn = wid - tm * ntn;
  int ln15 = lane & 15, khalf = (lane >> 4) * 8;

  const unsigned short* a0p = Ab + (long)(tm * 32 + ln15) * lda + khalf;
  const unsigned short* a1p = a0p + 16 * lda;
  const unsigned short* b0p = Bt + (long)(tn * 32 + ln15) * K + khalf;
  const unsigned short* b1p = b0p + 16 * (long)K;

  v8f c00 = {}, c01 = {}, c10 = {}, c11 = {};
  for (int k0 = 0; k0 < K; k0 += 32) {
    BF16x16 a0, a1, b0, b1;
    ldfrag(a0, a0p + k0); ldfrag(a1, a1p + k0);
    ldfrag(b0, b0p + k0); ldfrag(b1, b1p + k0);
    __builtin_prefetch((const void*)(a0p + k0 + 32), 0, 1);
    __builtin_prefetch((const void*)(b0p + k0 + 32), 0, 1);
    c00 = __builtin_amdgcn_wmma_f32_16x16x32_bf16(false, a0.v, false, b0.v, (short)0, c00, false, false);
    c01 = __builtin_amdgcn_wmma_f32_16x16x32_bf16(false, a0.v, false, b1.v, (short)0, c01, false, false);
    c10 = __builtin_amdgcn_wmma_f32_16x16x32_bf16(false, a1.v, false, b0.v, (short)0, c10, false, false);
    c11 = __builtin_amdgcn_wmma_f32_16x16x32_bf16(false, a1.v, false, b1.v, (short)0, c11, false, false);
  }
  int r0 = tm * 32 + khalf, cl0 = tn * 32 + ln15;
  store_tile(c00, r0,      cl0,      ldc, bias, Res, Cf, Cbf, relu);
  store_tile(c01, r0,      cl0 + 16, ldc, bias, Res, Cf, Cbf, relu);
  store_tile(c10, r0 + 16, cl0,      ldc, bias, Res, Cf, Cbf, relu);
  store_tile(c11, r0 + 16, cl0 + 16, ldc, bias, Res, Cf, Cbf, relu);
}

// ---------------- expert-FFN GEMM: TDM-staged LDS, 64x64 block tile, 4 waves ----------------
__global__ void dl_gemm_moe(const unsigned short* __restrict__ A, long lda, long aeS,
                            const unsigned short* __restrict__ Bt, long beS,
                            const float* __restrict__ bias, long biasS,
                            float* __restrict__ Cf, unsigned short* __restrict__ Cbf,
                            long ldc, long ceS, int M, int N, int K, int relu) {
  __shared__ __align__(16) unsigned short As[64 * 64];
  __shared__ __align__(16) unsigned short Bs[64 * 64];
  int tid = threadIdx.x;
  int wave = tid >> 5, lane = tid & 31;
  int ntn = N >> 6;
  int tm = blockIdx.x / ntn, tn = blockIdx.x - tm * ntn;
  long e = blockIdx.y;
  const unsigned short* Ae = A + e * aeS;
  const unsigned short* Be = Bt + e * beS;
  int m0 = tm * 64, n0 = tn * 64;
  int wr = (wave >> 1) & 1, wc = wave & 1;          // 2x2 waves, 32x32 each
  int ln15 = lane & 15, khalf = (lane >> 4) * 8;

  v8f c00 = {}, c01 = {}, c10 = {}, c11 = {};
  for (int k0 = 0; k0 < K; k0 += 64) {
#if HAVE_TDM
    if (wave == 0) {
      tdm_load_2d((unsigned)(unsigned long long)&As[0], Ae + (long)m0 * lda + k0,
                  64u, 64u, (unsigned)(K - k0), (unsigned)(M - m0), (unsigned long long)lda);
      tdm_load_2d((unsigned)(unsigned long long)&Bs[0], Be + (long)n0 * K + k0,
                  64u, 64u, (unsigned)(K - k0), (unsigned)(N - n0), (unsigned long long)K);
      __builtin_amdgcn_s_wait_tensorcnt(0);
    }
    __syncthreads();
#else
    for (int c = tid; c < 512; c += 128) {          // cooperative fallback copy
      int row = c >> 3, col = (c & 7) * 8;
      *(ushort8*)&As[row * 64 + col] = *(const ushort8*)&Ae[(long)(m0 + row) * lda + k0 + col];
      *(ushort8*)&Bs[row * 64 + col] = *(const ushort8*)&Be[(long)(n0 + row) * K + k0 + col];
    }
    __syncthreads();
#endif
#pragma unroll
    for (int kk = 0; kk < 64; kk += 32) {
      BF16x16 a0, a1, b0, b1;
      ldfrag(a0, &As[(wr * 32 + ln15) * 64 + kk + khalf]);
      ldfrag(a1, &As[(wr * 32 + 16 + ln15) * 64 + kk + khalf]);
      ldfrag(b0, &Bs[(wc * 32 + ln15) * 64 + kk + khalf]);
      ldfrag(b1, &Bs[(wc * 32 + 16 + ln15) * 64 + kk + khalf]);
      c00 = __builtin_amdgcn_wmma_f32_16x16x32_bf16(false, a0.v, false, b0.v, (short)0, c00, false, false);
      c01 = __builtin_amdgcn_wmma_f32_16x16x32_bf16(false, a0.v, false, b1.v, (short)0, c01, false, false);
      c10 = __builtin_amdgcn_wmma_f32_16x16x32_bf16(false, a1.v, false, b0.v, (short)0, c10, false, false);
      c11 = __builtin_amdgcn_wmma_f32_16x16x32_bf16(false, a1.v, false, b1.v, (short)0, c11, false, false);
    }
    __syncthreads();
  }
  const float* be_ = bias + e * biasS;
  float* cfe = Cf ? Cf + e * ceS : nullptr;
  unsigned short* cbe = Cbf ? Cbf + e * ceS : nullptr;
  int r0 = m0 + wr * 32 + khalf, cl0 = n0 + wc * 32 + ln15;
  store_tile(c00, r0,      cl0,      ldc, be_, nullptr, cfe, cbe, relu);
  store_tile(c01, r0,      cl0 + 16, ldc, be_, nullptr, cfe, cbe, relu);
  store_tile(c10, r0 + 16, cl0,      ldc, be_, nullptr, cfe, cbe, relu);
  store_tile(c11, r0 + 16, cl0 + 16, ldc, be_, nullptr, cfe, cbe, relu);
}

// ---------------- flash attention (one wave per 16-query tile per (b,h)) ----------------
__global__ void dl_attn(const unsigned short* __restrict__ Qb, int qStride,
                        const unsigned short* __restrict__ Kb,
                        const unsigned short* __restrict__ Vb, int kvStride,
                        unsigned short* __restrict__ Outb, int Tq, int Tk, int causal) {
  __shared__ __align__(16) unsigned short pls[16 * 32];
  int lane = threadIdx.x & 31;
  int bh = blockIdx.x, b = bh / Hh, h = bh % Hh;
  int qt0 = blockIdx.y * 16;
  int ln15 = lane & 15, khalf = (lane >> 4) * 8;

  BF16x16 qf0, qf1;
  {
    const unsigned short* qp = Qb + ((long)(b * Tq + qt0 + ln15)) * qStride + h * DH + khalf;
    ldfrag(qf0, qp);
    ldfrag(qf1, qp + 32);
  }
  float mrow[8], lrow[8];
  v8f O0 = {}, O1 = {}, O2 = {}, O3 = {};
#pragma unroll
  for (int r = 0; r < 8; ++r) { mrow[r] = -1e30f; lrow[r] = 0.f; }

  for (int s0 = 0; s0 < Tk; s0 += 32) {
    if (causal && s0 > qt0 + 15) break;            // wave-uniform early out
    v8f sc[2];
#pragma unroll
    for (int t = 0; t < 2; ++t) {
      int key = s0 + t * 16 + ln15;
      const unsigned short* kp = Kb + ((long)(b * Tk + key)) * kvStride + h * DH + khalf;
      BF16x16 k0f, k1f;
      ldfrag(k0f, kp);
      ldfrag(k1f, kp + 32);
      v8f c = {};
      c = __builtin_amdgcn_wmma_f32_16x16x32_bf16(false, qf0.v, false, k0f.v, (short)0, c, false, false);
      c = __builtin_amdgcn_wmma_f32_16x16x32_bf16(false, qf1.v, false, k1f.v, (short)0, c, false, false);
      sc[t] = c;
    }
    float mnew[8], sf[8];
#pragma unroll
    for (int r = 0; r < 8; ++r) {
      int qrow = qt0 + khalf + r;
      float a0 = sc[0][r] * 0.125f;                // 1/sqrt(64)
      float a1 = sc[1][r] * 0.125f;
      if (causal) {
        if (s0 + ln15 > qrow)      a0 = -1e9f;
        if (s0 + 16 + ln15 > qrow) a1 = -1e9f;
      }
      sc[0][r] = a0; sc[1][r] = a1;
      float rm = fmaxf(rowmax16(a0), rowmax16(a1));
      mnew[r] = fmaxf(mrow[r], rm);
      sf[r] = __expf(mrow[r] - mnew[r]);
      mrow[r] = mnew[r];
    }
#pragma unroll
    for (int r = 0; r < 8; ++r) {
      float p0 = __expf(sc[0][r] - mnew[r]);
      float p1 = __expf(sc[1][r] - mnew[r]);
      lrow[r] = lrow[r] * sf[r] + rowsum16(p0) + rowsum16(p1);
      O0[r] *= sf[r]; O1[r] *= sf[r]; O2[r] *= sf[r]; O3[r] *= sf[r];
      int row = khalf + r;
      pls[row * 32 + ln15]      = f2bf(p0);
      pls[row * 32 + 16 + ln15] = f2bf(p1);
    }
    __syncthreads();
    BF16x16 pf;                                    // P as A-matrix via LDS relayout
    ldfrag(pf, &pls[ln15 * 32 + khalf]);
    __syncthreads();
#pragma unroll
    for (int j = 0; j < 4; ++j) {
      BF16x16 vf;                                  // V as B-matrix 32x16
      int col = h * DH + j * 16 + ln15;
#pragma unroll
      for (int i = 0; i < 8; ++i) {
        int key0 = s0 + khalf + i;
        vf.s[i]     = Vb[((long)(b * Tk + key0)) * kvStride + col];
        vf.s[8 + i] = Vb[((long)(b * Tk + key0 + 16)) * kvStride + col];
      }
      v8f& Oj = (j == 0) ? O0 : (j == 1) ? O1 : (j == 2) ? O2 : O3;
      Oj = __builtin_amdgcn_wmma_f32_16x16x32_bf16(false, pf.v, false, vf.v, (short)0, Oj, false, false);
    }
  }
#pragma unroll
  for (int r = 0; r < 8; ++r) {
    long tok = (long)b * Tq + qt0 + khalf + r;
    float inv = 1.f / lrow[r];
    unsigned short* op = Outb + tok * Dd + h * DH;
    op[0 * 16 + ln15] = f2bf(O0[r] * inv);
    op[1 * 16 + ln15] = f2bf(O1[r] * inv);
    op[2 * 16 + ln15] = f2bf(O2[r] * inv);
    op[3 * 16 + ln15] = f2bf(O3[r] * inv);
  }
}

// ---------------- router (one wave per token) ----------------
__global__ void dl_router(const float* __restrict__ x, const float* __restrict__ rw,
                          const float* __restrict__ rb, const float* __restrict__ tmask,
                          float* probs, int* eidx, float* gate, float* lse, int* valid) {
  int w = (int)(((long)blockIdx.x * blockDim.x + threadIdx.x) >> 5);
  int lane = threadIdx.x & 31;
  if (w >= NTOK) return;
  const float* xr = x + (long)w * Dd;
  float acc[Ee];
#pragma unroll
  for (int e = 0; e < Ee; ++e) acc[e] = 0.f;
  for (int d = lane; d < Dd; d += 32) {
    float xv = xr[d];
    const float* rwp = rw + (long)d * Ee;
#pragma unroll
    for (int e = 0; e < Ee; ++e) acc[e] += xv * rwp[e];
  }
#pragma unroll
  for (int e = 0; e < Ee; ++e)
#pragma unroll
    for (int m = 1; m < 32; m <<= 1) acc[e] += __shfl_xor(acc[e], m, 32);
  if (lane == 0) {
    float lg[Ee], mx = -1e30f; int am = 0;
#pragma unroll
    for (int e = 0; e < Ee; ++e) { lg[e] = acc[e] + rb[e]; if (lg[e] > mx) { mx = lg[e]; am = e; } }
    float se = 0.f;
#pragma unroll
    for (int e = 0; e < Ee; ++e) se += __expf(lg[e] - mx);
#pragma unroll
    for (int e = 0; e < Ee; ++e) probs[(long)w * Ee + e] = __expf(lg[e] - mx) / se;
    eidx[w] = am;
    gate[w] = 1.f / se;                            // lg[am]==mx
    lse[w]  = mx + __logf(se);
    valid[w] = (tmask[w] > 0.f) ? 1 : 0;
  }
}

// ---------------- deterministic capacity scan + aux losses ----------------
__global__ void dl_scan(const float* __restrict__ probs, const int* __restrict__ eidx,
                        const int* __restrict__ valid, const float* __restrict__ lse,
                        int* pos, int* keep, float* out_tail) {
  if (threadIdx.x != 0 || blockIdx.x != 0) return;
  int counts[Ee]; float pe[Ee];
  for (int e = 0; e < Ee; ++e) { counts[e] = 0; pe[e] = 0.f; }
  float zsum = 0.f; int nv = 0;
  for (int n = 0; n < NTOK; ++n) {
    if (valid[n]) {
      ++nv; zsum += lse[n] * lse[n];
      for (int q = 0; q < Ee; ++q) pe[q] += probs[(long)n * Ee + q];
      int e = eidx[n];
      int p = counts[e]++;
      pos[n]  = (p < CAPc) ? p : (CAPc - 1);
      keep[n] = (p < CAPc) ? 1 : 0;
    } else { pos[n] = 0; keep[n] = 0; }
  }
  float fnv = (nv > 0) ? (float)nv : 1.f;
  float lb = 0.f;
  for (int e = 0; e < Ee; ++e) lb += ((float)counts[e] / fnv) * (pe[e] / fnv);
  out_tail[0] = (float)Ee * lb;
  out_tail[1] = zsum / fnv;
}

// ---------------- MoE dispatch / combine ----------------
__global__ void dl_dispatch(const float* __restrict__ tln, const int* __restrict__ eidx,
                            const int* __restrict__ pos, const int* __restrict__ keep,
                            unsigned short* __restrict__ buf) {
  int n = blockIdx.x, d = blockIdx.y * 256 + threadIdx.x;
  if (!keep[n]) return;
  long slot = ((long)eidx[n] * CAPc + pos[n]) * Dd + d;
  buf[slot] = f2bf(tln[(long)n * Dd + d]);
}

__global__ void dl_combine(const float* __restrict__ xres, const float* __restrict__ ob,
                           const int* __restrict__ eidx, const int* __restrict__ pos,
                           const int* __restrict__ keep, const float* __restrict__ gate,
                           float* __restrict__ out) {
  int n = blockIdx.x, d = blockIdx.y * 256 + threadIdx.x;
  long i = (long)n * Dd + d;
  float y = 0.f;
  if (keep[n]) y = ob[((long)eidx[n] * CAPc + pos[n]) * Dd + d] * gate[n];
  out[i] = xres[i] + y;
}

// ---------------- workspace layout (bytes) ----------------
constexpr size_t OFF_XRES   = 0;                          // f32  [4096][1024]
constexpr size_t OFF_TLN    = OFF_XRES   + 16777216;      // f32  [4096][1024]
constexpr size_t OFF_TLNB   = OFF_TLN    + 16777216;      // bf16 [4096][1024]
constexpr size_t OFF_SRCB   = OFF_TLNB   + 8388608;       // bf16 [4096][1024]
constexpr size_t OFF_QKVB   = OFF_SRCB   + 8388608;       // bf16 [4096][3072] (cross: q + kv)
constexpr size_t OFF_CTXB   = OFF_QKVB   + 25165824;      // bf16 [4096][1024]
constexpr size_t OFF_WIN_S  = OFF_CTXB   + 8388608;       // bf16 [3072][1024]
constexpr size_t OFF_WOUT_S = OFF_WIN_S  + 6291456;       // bf16 [1024][1024]
constexpr size_t OFF_WIN_E  = OFF_WOUT_S + 2097152;
constexpr size_t OFF_WOUT_E = OFF_WIN_E  + 6291456;
constexpr size_t OFF_W1T    = OFF_WOUT_E + 2097152;       // bf16 [E][F][D]
constexpr size_t OFF_W2T    = OFF_W1T    + 67108864;      // bf16 [E][D][F]
constexpr size_t OFF_PROBS  = OFF_W2T    + 67108864;      // f32 [4096][8]
constexpr size_t OFF_EIDX   = OFF_PROBS  + 131072;
constexpr size_t OFF_GATE   = OFF_EIDX   + 16384;
constexpr size_t OFF_LSE    = OFF_GATE   + 16384;
constexpr size_t OFF_VALID  = OFF_LSE    + 16384;
constexpr size_t OFF_POS    = OFF_VALID  + 16384;
constexpr size_t OFF_KEEP   = OFF_POS    + 16384;
constexpr size_t OFF_BUF    = OFF_KEEP   + 16384;         // bf16 [E][CAP][D]
constexpr size_t OFF_HH     = OFF_BUF    + 10485760;      // bf16 [E][CAP][F]
constexpr size_t OFF_OB     = OFF_HH     + 41943040;      // f32  [E][CAP][D]

extern "C" void kernel_launch(void* const* d_in, const int* in_sizes, int n_in,
                              void* d_out, int out_size, void* d_ws, size_t ws_size,
                              hipStream_t stream) {
  (void)in_sizes; (void)n_in; (void)out_size; (void)ws_size;
  const float* tgt        = (const float*)d_in[0];
  const float* src        = (const float*)d_in[1];
  const float* token_mask = (const float*)d_in[5];
  const float* ln1_w = (const float*)d_in[6],  *ln1_b = (const float*)d_in[7];
  const float* self_in_w  = (const float*)d_in[8],  *self_in_b  = (const float*)d_in[9];
  const float* self_out_w = (const float*)d_in[10], *self_out_b = (const float*)d_in[11];
  const float* ln2_w = (const float*)d_in[12], *ln2_b = (const float*)d_in[13];
  const float* enc_in_w   = (const float*)d_in[14], *enc_in_b   = (const float*)d_in[15];
  const float* enc_out_w  = (const float*)d_in[16], *enc_out_b  = (const float*)d_in[17];
  const float* ln3_w = (const float*)d_in[18], *ln3_b = (const float*)d_in[19];
  const float* router_w = (const float*)d_in[20], *router_b = (const float*)d_in[21];
  const float* expert_w1 = (const float*)d_in[22], *expert_b1 = (const float*)d_in[23];
  const float* expert_w2 = (const float*)d_in[24], *expert_b2 = (const float*)d_in[25];

  char* ws = (char*)d_ws;
  float*          xres  = (float*)(ws + OFF_XRES);
  float*          tln   = (float*)(ws + OFF_TLN);
  unsigned short* tlnb  = (unsigned short*)(ws + OFF_TLNB);
  unsigned short* srcb  = (unsigned short*)(ws + OFF_SRCB);
  unsigned short* qkvb  = (unsigned short*)(ws + OFF_QKVB);
  unsigned short* ctxb  = (unsigned short*)(ws + OFF_CTXB);
  unsigned short* win_s = (unsigned short*)(ws + OFF_WIN_S);
  unsigned short* wout_s= (unsigned short*)(ws + OFF_WOUT_S);
  unsigned short* win_e = (unsigned short*)(ws + OFF_WIN_E);
  unsigned short* wout_e= (unsigned short*)(ws + OFF_WOUT_E);
  unsigned short* w1t   = (unsigned short*)(ws + OFF_W1T);
  unsigned short* w2t   = (unsigned short*)(ws + OFF_W2T);
  float* probs = (float*)(ws + OFF_PROBS);
  int*   eidx  = (int*)(ws + OFF_EIDX);
  float* gate  = (float*)(ws + OFF_GATE);
  float* lse   = (float*)(ws + OFF_LSE);
  int*   valid = (int*)(ws + OFF_VALID);
  int*   pos   = (int*)(ws + OFF_POS);
  int*   keep  = (int*)(ws + OFF_KEEP);
  unsigned short* bufb = (unsigned short*)(ws + OFF_BUF);
  unsigned short* hh   = (unsigned short*)(ws + OFF_HH);
  float* ob    = (float*)(ws + OFF_OB);
  float* out   = (float*)d_out;

  auto gemm = [&](const unsigned short* Ab, long lda, const unsigned short* Bt,
                  const float* bias, float* Cf, unsigned short* Cbf, long ldc,
                  const float* Res, int M, int N, int K, int relu) {
    int tiles = (M / 32) * (N / 32);
    dl_gemm_wmma<<<dim3((tiles + 7) / 8), dim3(256), 0, stream>>>(
        Ab, lda, Bt, bias, Cf, Cbf, ldc, Res, M, N, K, relu);
  };

  // ---- weight prep (bf16 conversion / per-expert transpose) ----
  dl_conv_bf16<<<(3 * Dd * Dd + 255) / 256, 256, 0, stream>>>(self_in_w,  win_s, 3L * Dd * Dd);
  dl_conv_bf16<<<(Dd * Dd + 255) / 256,     256, 0, stream>>>(self_out_w, wout_s, (long)Dd * Dd);
  dl_conv_bf16<<<(3 * Dd * Dd + 255) / 256, 256, 0, stream>>>(enc_in_w,   win_e, 3L * Dd * Dd);
  dl_conv_bf16<<<(Dd * Dd + 255) / 256,     256, 0, stream>>>(enc_out_w,  wout_e, (long)Dd * Dd);
  dl_conv_bf16<<<(NTOK * Dd + 255) / 256,   256, 0, stream>>>(src, srcb, (long)NTOK * Dd);
  dl_transconv<<<dim3(Ff / 32, Dd / 32, Ee), 256, 0, stream>>>(expert_w1, w1t, Dd, Ff); // [D][F]->[F][D]
  dl_transconv<<<dim3(Dd / 32, Ff / 32, Ee), 256, 0, stream>>>(expert_w2, w2t, Ff, Dd); // [F][D]->[D][F]

  // ---- block 1: LN1 + causal self-attention + residual ----
  dl_ln<<<NTOK, 256, 0, stream>>>(tgt, ln1_w, ln1_b, tln, tlnb);
  gemm(tlnb, Dd, win_s, self_in_b, nullptr, qkvb, 3 * Dd, nullptr, NTOK, 3 * Dd, Dd, 0);
  dl_attn<<<dim3(Bb * Hh, Tt / 16), 32, 0, stream>>>(qkvb, 3 * Dd,
                                                     qkvb + Dd, qkvb + 2 * Dd, 3 * Dd,
                                                     ctxb, Tt, Tt, 1);
  gemm(ctxb, Dd, wout_s, self_out_b, xres, nullptr, Dd, tgt, NTOK, Dd, Dd, 0);

  // ---- block 2: LN2 + cross-attention + residual ----
  dl_ln<<<NTOK, 256, 0, stream>>>(xres, ln2_w, ln2_b, tln, tlnb);
  gemm(tlnb, Dd, win_e, enc_in_b, nullptr, qkvb, Dd, nullptr, NTOK, Dd, Dd, 0);       // q
  unsigned short* kvb = qkvb + (long)NTOK * Dd;                                       // bf16 [4096][2048]
  gemm(srcb, Dd, win_e + (long)Dd * Dd, enc_in_b + Dd, nullptr, kvb, 2 * Dd, nullptr,
       Bb * Ss, 2 * Dd, Dd, 0);                                                       // k,v
  dl_attn<<<dim3(Bb * Hh, Tt / 16), 32, 0, stream>>>(qkvb, Dd,
                                                     kvb, kvb + Dd, 2 * Dd,
                                                     ctxb, Tt, Ss, 0);
  gemm(ctxb, Dd, wout_e, enc_out_b, xres, nullptr, Dd, xres, NTOK, Dd, Dd, 0);

  // ---- block 3: LN3 + Switch MoE ----
  dl_ln<<<NTOK, 256, 0, stream>>>(xres, ln3_w, ln3_b, tln, tlnb);
  dl_router<<<(NTOK * 32) / 256, 256, 0, stream>>>(tln, router_w, router_b, token_mask,
                                                   probs, eidx, gate, lse, valid);
  dl_scan<<<1, 1, 0, stream>>>(probs, eidx, valid, lse, pos, keep, out + (long)NTOK * Dd);
  hipMemsetAsync(bufb, 0, (size_t)Ee * CAPc * Dd * 2, stream);
  dl_dispatch<<<dim3(NTOK, Dd / 256), 256, 0, stream>>>(tln, eidx, pos, keep, bufb);
  dl_gemm_moe<<<dim3((CAPc / 64) * (Ff / 64), Ee), 128, 0, stream>>>(
      bufb, Dd, (long)CAPc * Dd, w1t, (long)Ff * Dd, expert_b1, Ff,
      nullptr, hh, Ff, (long)CAPc * Ff, CAPc, Ff, Dd, 1);
  dl_gemm_moe<<<dim3((CAPc / 64) * (Dd / 64), Ee), 128, 0, stream>>>(
      hh, Ff, (long)CAPc * Ff, w2t, (long)Dd * Ff, expert_b2, Dd,
      ob, nullptr, Dd, (long)CAPc * Dd, CAPc, Dd, Ff, 0);
  dl_combine<<<dim3(NTOK, Dd / 256), 256, 0, stream>>>(xres, ob, eidx, pos, keep, gate, out);
}
